// KeyframeNet_18176301596758
// MI455X (gfx1250) — matrix operand order repaired and checked by reading, not compiled, not run on hardware
//
#include <hip/hip_runtime.h>
#include <hip/hip_bf16.h>

#define BB   16
#define TT   512
#define DM   512
#define NH   8
#define DH   64
#define DFF  2048
#define CTXN 10
#define BTN  (BB*TT)      /* 8192 tokens */
#define SCALE_V 0.125f    /* 1/sqrt(64) */

typedef __attribute__((ext_vector_type(16))) __bf16 v16bf;
typedef __attribute__((ext_vector_type(8)))  __bf16 v8bf;
typedef __attribute__((ext_vector_type(8)))  float  v8f;

__device__ __forceinline__ __bf16 f2bf(float x) {
  unsigned u = __builtin_bit_cast(unsigned, x);
  unsigned r = u + 0x7FFFu + ((u >> 16) & 1u);   // round-to-nearest-even
  unsigned short s = (unsigned short)(r >> 16);
  return __builtin_bit_cast(__bf16, s);
}

// ---------------------------------------------------------------------------
// Batched bf16 WMMA GEMM: C[M,N] = act(A[M,Kp] @ B[Kp,N] + bias) (+ residual)
//  A  : bf16 row-major, lda mult of 8, K-pad zero-filled, Kp mult of 32
//  Bt : bf16 N x Kp (B transposed), ldbt mult of 8, zero-filled pad
//  C  : optional f32 out; Cb: optional bf16 out (same ldc)
//  M/N edges: loads CLAMP to last valid row/col (results discarded at store)
//  diagT>0: rel-pos scatter mode: out(i,r) += C[i][r-(diagT-1)+i]
//  Wave computes a 32x64 tile: per 32-k step, 2 A frags + 4 B frags are
//  pre-loaded into distinct registers (one load clause, one wait), then
//  8x v_wmma_f32_16x16x32_bf16.
// ---------------------------------------------------------------------------
__global__ __launch_bounds__(128) void gemm_bf16_kernel(
    const __bf16* __restrict__ A, int lda, long long sAb,
    const __bf16* __restrict__ Bt, int ldbt, long long sBb,
    float* __restrict__ C, int ldc, long long sCb,
    __bf16* __restrict__ Cb, long long sCbb,
    const float* __restrict__ bias,
    const float* __restrict__ Res,
    const float* __restrict__ alphaPtr, int act,
    int M, int N, int Kp, int diagT)
{
  const int lane = threadIdx.x & 31;
  const int wave = threadIdx.x >> 5;
  const int tilesM = (M + 31) >> 5;
  const int tilesN = (N + 63) >> 6;
  long long tile = (long long)blockIdx.x * 4 + wave;
  if (tile >= (long long)tilesM * tilesN) return;
  const int tm = (int)(tile % tilesM);
  const int tn = (int)(tile / tilesM);
  const long long bI = blockIdx.y;
  A  += bI * sAb;
  Bt += bI * sBb;
  if (C)   C   += bI * sCb;
  if (Cb)  Cb  += bI * sCbb;
  if (Res) Res += bI * sCb;

  const int row0 = tm << 5;
  const int col0 = tn << 6;
  const int lid  = lane & 15;
  const int hi   = lane >> 4;
  const int koffA = hi << 3;            // A: hi lanes hold K {8..15, 24..31}
  const int koffB = hi << 3;            // B: hi lanes hold K {8..15, 24..31}

  int am0 = row0 + lid;      if (am0 > M - 1) am0 = M - 1;   // clamped rows
  int am1 = row0 + 16 + lid; if (am1 > M - 1) am1 = M - 1;
  const __bf16* ap0 = A + (long long)am0 * lda;
  const __bf16* ap1 = A + (long long)am1 * lda;
  const __bf16* bp[4];
#pragma unroll
  for (int t = 0; t < 4; t++) {
    int bn = col0 + (t << 4) + lid; if (bn > N - 1) bn = N - 1;
    bp[t] = Bt + (long long)bn * ldbt;
  }

  v8f acc[2][4];
#pragma unroll
  for (int s = 0; s < 2; s++)
#pragma unroll
    for (int t = 0; t < 4; t++)
#pragma unroll
      for (int r = 0; r < 8; r++) acc[s][t][r] = 0.0f;

  union FR { v16bf f; v8bf h[2]; };

  for (int k0 = 0; k0 < Kp; k0 += 32) {
    FR fa0, fa1, fb[4];
    fa0.h[0] = *(const v8bf*)(ap0 + k0 + koffA);
    fa0.h[1] = *(const v8bf*)(ap0 + k0 + koffA + 16);
    fa1.h[0] = *(const v8bf*)(ap1 + k0 + koffA);
    fa1.h[1] = *(const v8bf*)(ap1 + k0 + koffA + 16);
#pragma unroll
    for (int t = 0; t < 4; t++) {
      fb[t].h[0] = *(const v8bf*)(bp[t] + k0 + koffB);
      fb[t].h[1] = *(const v8bf*)(bp[t] + k0 + koffB + 16);
    }
#pragma unroll
    for (int t = 0; t < 4; t++)
      acc[0][t] = __builtin_amdgcn_wmma_f32_16x16x32_bf16(
          false, fa0.f, false, fb[t].f, (short)0, acc[0][t], false, false);
#pragma unroll
    for (int t = 0; t < 4; t++)
      acc[1][t] = __builtin_amdgcn_wmma_f32_16x16x32_bf16(
          false, fa1.f, false, fb[t].f, (short)0, acc[1][t], false, false);
  }

  const float alpha = alphaPtr ? alphaPtr[0] : 0.0f;
  const int rbase = hi << 3;            // C: lane = N col, vgpr r = M row (+8 hi)
  if (diagT > 0) {
#pragma unroll
    for (int s = 0; s < 2; s++)
#pragma unroll
      for (int t = 0; t < 4; t++) {
        const int cn = col0 + (t << 4) + lid;
#pragma unroll
        for (int r = 0; r < 8; r++) {
          const int cm = row0 + (s << 4) + rbase + r;
          if (cm < M && cn < N) {
            int j = cn - (diagT - 1) + cm;          // unique (i,j) target
            if (j >= 0 && j < diagT) C[(long long)cm * ldc + j] += acc[s][t][r];
          }
        }
      }
    return;
  }
  const bool full = (row0 + 32 <= M) && (col0 + 64 <= N);
#pragma unroll
  for (int s = 0; s < 2; s++)
#pragma unroll
    for (int t = 0; t < 4; t++) {
      const int cn = col0 + (t << 4) + lid;
#pragma unroll
      for (int r = 0; r < 8; r++) {
        const int cm = row0 + (s << 4) + rbase + r;
        if (!full && (cm >= M || cn >= N)) continue;
        float val = acc[s][t][r];
        if (bias) val += bias[cn];
        if (act == 1) val = fmaxf(val, 0.0f);                      // relu
        else if (act == 2) val = (val >= 0.0f) ? val : alpha*val;  // prelu
        const long long o = (long long)cm * ldc + cn;
        if (Res) val += Res[o];
        if (C)  C[o]  = val;
        if (Cb) Cb[o] = f2bf(val);
      }
    }
}

// ------------------------- elementwise / reduction kernels -----------------
__global__ __launch_bounds__(256) void ln_kernel(
    const float* __restrict__ X, __bf16* __restrict__ Y,
    const float* __restrict__ g, const float* __restrict__ b, int rows)
{
  int row = blockIdx.x * 8 + (threadIdx.x >> 5);
  int lane = threadIdx.x & 31;
  if (row >= rows) return;
  const float* xp = X + (long long)row * DM;
  float vals[16]; float s = 0.f, s2 = 0.f;
#pragma unroll
  for (int i = 0; i < 16; i++) { float v = xp[lane + i*32]; vals[i]=v; s+=v; s2+=v*v; }
#pragma unroll
  for (int o = 16; o > 0; o >>= 1) { s += __shfl_xor(s,o,32); s2 += __shfl_xor(s2,o,32); }
  float mean = s * (1.0f/DM);
  float var  = s2 * (1.0f/DM) - mean*mean;
  float inv  = rsqrtf(var + 1e-5f);
  __bf16* yp = Y + (long long)row * DM;
#pragma unroll
  for (int i = 0; i < 16; i++) { int c = lane + i*32; yp[c] = f2bf((vals[i]-mean)*inv*g[c] + b[c]); }
}

__global__ __launch_bounds__(256) void softmax_kernel(
    const float* __restrict__ S, __bf16* __restrict__ O, int rows, int useMask)
{
  int row = blockIdx.x * 8 + (threadIdx.x >> 5);
  int lane = threadIdx.x & 31;
  if (row >= rows) return;
  const float* p = S + (long long)row * TT;
  float v[16]; float mx = -3.0e38f;
#pragma unroll
  for (int i = 0; i < 16; i++) {
    int j = lane + i*32;
    float a = p[j] * SCALE_V;
    if (useMask && (j >= CTXN) && (j < TT-1)) a = -1.0e9f;
    v[i] = a; mx = fmaxf(mx, a);
  }
#pragma unroll
  for (int o = 16; o > 0; o >>= 1) mx = fmaxf(mx, __shfl_xor(mx,o,32));
  float sum = 0.f;
#pragma unroll
  for (int i = 0; i < 16; i++) { float e = __expf(v[i]-mx); v[i]=e; sum+=e; }
#pragma unroll
  for (int o = 16; o > 0; o >>= 1) sum += __shfl_xor(sum,o,32);
  float inv = 1.0f / sum;
  __bf16* op = O + (long long)row * TT;
#pragma unroll
  for (int i = 0; i < 16; i++) op[lane + i*32] = f2bf(v[i]*inv);
}

__global__ void prep_weight_kernel(const float* __restrict__ W,
                                   __bf16* __restrict__ Wt, int K, int N, int ldp)
{
  long long idx = (long long)blockIdx.x * blockDim.x + threadIdx.x;
  if (idx >= (long long)N * ldp) return;
  int k = (int)(idx % ldp);
  long long n = idx / ldp;
  Wt[idx] = (k < K) ? f2bf(W[(long long)k * N + n]) : f2bf(0.0f);
}

__global__ void f32_to_bf16_kernel(const float* __restrict__ s,
                                   __bf16* __restrict__ d, long long n)
{
  long long i = (long long)blockIdx.x * blockDim.x + threadIdx.x;
  if (i < n) d[i] = f2bf(s[i]);
}

__global__ void transpose_v_kernel(const __bf16* __restrict__ V, __bf16* __restrict__ Vt)
{
  long long idx = (long long)blockIdx.x * blockDim.x + threadIdx.x;
  if (idx >= (long long)BB*NH*DH*TT) return;
  int j  = (int)(idx & (TT-1));
  long long r = idx >> 9;
  int d  = (int)(r & (DH-1)); r >>= 6;
  int hh = (int)(r & (NH-1));
  int b  = (int)(r >> 3);
  Vt[idx] = V[(((long long)b*TT + j)*NH + hh)*DH + d];
}

__global__ void build_enc_in_kernel(const float* __restrict__ motion, __bf16* __restrict__ E)
{
  long long idx = (long long)blockIdx.x * blockDim.x + threadIdx.x;
  if (idx >= (long long)BTN * 288) return;
  int c = (int)(idx % 288);
  long long bt = idx / 288;
  int t = (int)(bt & (TT-1));
  float bm = (t >= CTXN && t < TT-1) ? 0.0f : 1.0f;
  float v = 0.0f;
  if (c < 256)       v = motion[bt*256 + c] * bm;
  else if (c == 256) v = bm;
  E[idx] = f2bf(v);
}

__global__ void build_kf_in_kernel(__bf16* __restrict__ KFI)
{
  int idx = blockIdx.x * blockDim.x + threadIdx.x;
  if (idx >= TT*32) return;
  int c = idx & 31, t = idx >> 5;
  float v = (c == 0) ? ((float)t - (float)(CTXN-1))
          : (c == 1) ? ((float)t - (float)(TT-1)) : 0.0f;
  KFI[idx] = f2bf(v);
}

__global__ void build_lt_kernel(__bf16* __restrict__ L)
{
  int idx = blockIdx.x * blockDim.x + threadIdx.x;
  if (idx >= (2*TT-1)*32) return;
  int c = idx & 31, r = idx >> 5;
  L[idx] = f2bf((c == 0) ? (float)(r - (TT-1)) : 0.0f);
}

__global__ void add_kf_kernel(float* __restrict__ X, const float* __restrict__ KF)
{
  long long idx = (long long)blockIdx.x * blockDim.x + threadIdx.x;
  if (idx >= (long long)BTN * DM) return;
  int c = (int)(idx & (DM-1));
  int t = (int)((idx >> 9) & (TT-1));
  X[idx] += KF[t*DM + c];
}

__global__ void finalize_kernel(const float* __restrict__ dec,
                                const float* __restrict__ motion,
                                float* __restrict__ mout, float* __restrict__ score)
{
  long long idx = (long long)blockIdx.x * blockDim.x + threadIdx.x;
  if (idx >= (long long)BTN * 257) return;
  int c = (int)(idx % 257);
  long long bt = idx / 257;
  int t = (int)(bt & (TT-1));
  float v = dec[idx];
  if (c < 256) {
    bool bm = (t < CTXN) || (t == TT-1);
    mout[bt*256 + c] = bm ? motion[bt*256 + c] : v;
  } else {
    score[bt] = 1.0f / (1.0f + __expf(-v));
  }
}

// ------------------------------- host side ---------------------------------
static inline int blk(long long n) { return (int)((n + 255) / 256); }

struct WT { __bf16* p; int ld; };

static void launch_gemm(hipStream_t s,
    const __bf16* A, int lda, long long sAb,
    const __bf16* Bt, int ldbt, long long sBb,
    float* C, int ldc, long long sCb,
    __bf16* Cb, long long sCbb,
    const float* bias, const float* Res, const float* alpha, int act,
    int M, int N, int Kp, int batches, int diagT)
{
  int tilesM = (M + 31) / 32;
  int tilesN = (N + 63) / 64;
  long long waves = (long long)tilesM * tilesN;
  dim3 grid((unsigned)((waves + 3) / 4), (unsigned)batches, 1);
  gemm_bf16_kernel<<<grid, 128, 0, s>>>(A, lda, sAb, Bt, ldbt, sBb, C, ldc, sCb,
                                        Cb, sCbb, bias, Res, alpha, act, M, N, Kp, diagT);
}

extern "C" void kernel_launch(void* const* d_in, const int* in_sizes, int n_in,
                              void* d_out, int out_size, void* d_ws, size_t ws_size,
                              hipStream_t stream)
{
  (void)in_sizes; (void)n_in; (void)out_size; (void)ws_size;
  const float* motion = (const float*)d_in[0];
  const float* trajp  = (const float*)d_in[1];
  auto P = [&](int i) -> const float* { return (const float*)d_in[i]; };

  // ---- workspace allocator (deterministic, ~175 MB) ----
  char* base = (char*)d_ws; size_t off = 0;
  auto allocB = [&](size_t bytes) -> void* {
    void* p = base + off; off = (off + bytes + 255) & ~((size_t)255); return p; };
  auto aF = [&](size_t n) -> float*  { return (float*)allocB(n * 4); };
  auto aH = [&](size_t n) -> __bf16* { return (__bf16*)allocB(n * 2); };

  float*  x    = aF((size_t)BTN*DM);          // residual stream (f32)
  float*  ctx  = aF((size_t)BTN*DM);
  float*  sc   = aF((size_t)NH*TT*TT);        // raw attention scores (per b)
  float*  kfb  = aF((size_t)TT*DM);
  float*  dec  = aF((size_t)BTN*257);
  __bf16* h    = aH((size_t)BTN*DM);          // LN outputs (GEMM A side)
  __bf16* hc   = aH((size_t)BTN*DM);
  __bf16* qb   = aH((size_t)BTN*DM);
  __bf16* kb   = aH((size_t)BTN*DM);
  __bf16* vb   = aH((size_t)BTN*DM);
  __bf16* vt   = aH((size_t)BTN*DM);          // V transposed (B,H,D,T)
  __bf16* tb   = aH((size_t)BTN*DM);          // generic bf16 hidden / attn-o
  __bf16* scb  = aH((size_t)NH*TT*TT);        // softmax probs (per b)
  __bf16* ffhb = aH((size_t)TT*DFF);          // FFN hidden (per b)
  __bf16* enc  = aH((size_t)BTN*288);         // motion concat, K padded 257->288
  __bf16* trb  = aH((size_t)BTN*32);          // traj in bf16
  __bf16* kfin = aH((size_t)TT*32);           // kf in, K padded 2->32
  __bf16* ltb  = aH((size_t)(2*TT-1)*32);     // rel-pos in, K padded 1->32
  __bf16* lookb= aH((size_t)(2*TT-1)*DH);     // rel-pos lookup (B-side of qe)

  // ---- one-time weight convert+transpose to bf16 (N x round32(K)) ----
  auto prep = [&](int wi, int K, int N) -> WT {
    int ldp = (K + 31) & ~31;
    __bf16* wt = aH((size_t)N * ldp);
    prep_weight_kernel<<<blk((long long)N * ldp), 256, 0, stream>>>(P(wi), wt, K, N, ldp);
    WT r; r.p = wt; r.ld = ldp; return r;
  };

  WT meW1 = prep(2, 257, DM),  meW2 = prep(5, DM, DM);
  WT teW1 = prep(8, 32, DM),   teW2 = prep(11, DM, DM);
  WT kfW1 = prep(14, 2, DM),   kfW2 = prep(17, DM, DM);
  WT rpW1 = prep(19, 1, DM),   rpW2 = prep(22, DM, DH);
  WT aWq[6], aWk[6], aWv[6], aWo[6], cWq[6], cWk[6], cWv[6], cWo[6], fW1[6], fW2[6];
  for (int l = 0; l < 6; l++) {
    int bA = 24 + l*26;
    aWq[l]=prep(bA+2,DM,DM);  aWk[l]=prep(bA+4,DM,DM);
    aWv[l]=prep(bA+6,DM,DM);  aWo[l]=prep(bA+8,DM,DM);
    int bC = bA + 10;
    cWq[l]=prep(bC+2,DM,DM);  cWk[l]=prep(bC+4,DM,DM);
    cWv[l]=prep(bC+6,DM,DM);  cWo[l]=prep(bC+8,DM,DM);
    int bF = bA + 20;
    fW1[l]=prep(bF+2,DM,DFF); fW2[l]=prep(bF+4,DFF,DM);
  }
  WT dcW1 = prep(182, DM, DM), dcW2 = prep(185, DM, 257);

  // ---- encoders ----
  build_enc_in_kernel<<<blk((long long)BTN*288), 256, 0, stream>>>(motion, enc);
  f32_to_bf16_kernel<<<blk((long long)BTN*32), 256, 0, stream>>>(trajp, trb, (long long)BTN*32);
  launch_gemm(stream, enc,288,0, meW1.p,meW1.ld,0, nullptr,DM,0, tb,0, P(3),nullptr,P(4),2,  BTN,DM,288,1,0);
  launch_gemm(stream, tb, DM,0,  meW2.p,meW2.ld,0, x,     DM,0, nullptr,0, P(6),nullptr,P(7),2,  BTN,DM,DM, 1,0);
  launch_gemm(stream, trb,32,0,  teW1.p,teW1.ld,0, nullptr,DM,0, tb,0, P(9),nullptr,P(10),2, BTN,DM,32, 1,0);
  launch_gemm(stream, tb, DM,0,  teW2.p,teW2.ld,0, ctx,   DM,0, nullptr,0, P(12),nullptr,P(13),2,BTN,DM,DM, 1,0);
  build_kf_in_kernel<<<blk(TT*32), 256, 0, stream>>>(kfin);
  launch_gemm(stream, kfin,32,0, kfW1.p,kfW1.ld,0, nullptr,DM,0, tb,0, P(15),nullptr,P(16),2, TT,DM,32, 1,0);
  launch_gemm(stream, tb, DM,0,  kfW2.p,kfW2.ld,0, kfb,   DM,0, nullptr,0, P(18),nullptr,nullptr,0, TT,DM,DM,1,0);
  add_kf_kernel<<<blk((long long)BTN*DM), 256, 0, stream>>>(x, kfb);
  add_kf_kernel<<<blk((long long)BTN*DM), 256, 0, stream>>>(ctx, kfb);
  build_lt_kernel<<<blk((2*TT-1)*32), 256, 0, stream>>>(ltb);
  launch_gemm(stream, ltb,32,0,  rpW1.p,rpW1.ld,0, nullptr,DM,0, tb,0, P(20),nullptr,P(21),2, 2*TT-1,DM,32,1,0);
  launch_gemm(stream, tb, DM,0,  rpW2.p,rpW2.ld,0, nullptr,DH,0, lookb,0, P(23),nullptr,nullptr,0, 2*TT-1,DH,DM,1,0);

  // ---- transformer layers ----
  auto runMHA = [&](int l, bool cross) {
    int bA = 24 + l*26 + (cross ? 10 : 0);
    const float* lng = P(bA+0); const float* lnb = P(bA+1);
    const float* bq = P(bA+3), *bk = P(bA+5), *bv = P(bA+7), *bo = P(bA+9);
    WT Wq = cross ? cWq[l] : aWq[l];  WT Wk = cross ? cWk[l] : aWk[l];
    WT Wv = cross ? cWv[l] : aWv[l];  WT Wo = cross ? cWo[l] : aWo[l];

    ln_kernel<<<(BTN+7)/8, 256, 0, stream>>>(x, h, lng, lnb, BTN);
    const __bf16* hk = h;
    if (cross) { ln_kernel<<<(BTN+7)/8, 256, 0, stream>>>(ctx, hc, lng, lnb, BTN); hk = hc; }

    launch_gemm(stream, h, DM,0, Wq.p,Wq.ld,0, nullptr,DM,0, qb,0, bq,nullptr,nullptr,0, BTN,DM,DM,1,0);
    launch_gemm(stream, hk,DM,0, Wk.p,Wk.ld,0, nullptr,DM,0, kb,0, bk,nullptr,nullptr,0, BTN,DM,DM,1,0);
    launch_gemm(stream, hk,DM,0, Wv.p,Wv.ld,0, nullptr,DM,0, vb,0, bv,nullptr,nullptr,0, BTN,DM,DM,1,0);
    transpose_v_kernel<<<blk((long long)BTN*DM), 256, 0, stream>>>(vb, vt);

    for (int b = 0; b < BB; b++) {
      const __bf16* qB  = qb + (size_t)b*TT*DM;
      const __bf16* kB  = kb + (size_t)b*TT*DM;
      const __bf16* vtB = vt + (size_t)b*NH*DH*TT;
      // scores[h,i,j] = q . k   (batched over heads via gridDim.y)
      launch_gemm(stream, qB,DM,64, kB,DM,64, sc,TT,(long long)TT*TT, nullptr,0,
                  nullptr,nullptr,nullptr,0, TT,TT,DH, NH, 0);
      // rel-pos bias (q @ lookup^T) scatter-added along diagonals
      launch_gemm(stream, qB,DM,64, lookb,DH,0, sc,TT,(long long)TT*TT, nullptr,0,
                  nullptr,nullptr,nullptr,0, TT,2*TT-1,DH, NH, TT);
      softmax_kernel<<<(NH*TT+7)/8, 256, 0, stream>>>(sc, scb, NH*TT, cross ? 0 : 1);
      // O = P @ V  -> bf16 attn-o directly
      launch_gemm(stream, scb,TT,(long long)TT*TT, vtB,TT,(long long)DH*TT,
                  nullptr, DM, 0, tb + (size_t)b*TT*DM, 64,
                  nullptr,nullptr,nullptr,0, TT,DH,TT, NH, 0);
    }
    launch_gemm(stream, tb,DM,0, Wo.p,Wo.ld,0, x,DM,0, nullptr,0, bo, x, nullptr,0, BTN,DM,DM,1,0);
  };

  auto runFFN = [&](int l) {
    int bF = 24 + l*26 + 20;
    ln_kernel<<<(BTN+7)/8, 256, 0, stream>>>(x, h, P(bF+0), P(bF+1), BTN);
    for (int b = 0; b < BB; b++) {
      const __bf16* hb = h + (size_t)b*TT*DM;
      float* xb = x + (size_t)b*TT*DM;
      launch_gemm(stream, hb,DM,0, fW1[l].p,fW1[l].ld,0, nullptr,DFF,0, ffhb,0,
                  P(bF+3),nullptr,nullptr,1, TT,DFF,DM,1,0);
      launch_gemm(stream, ffhb,DFF,0, fW2[l].p,fW2[l].ld,0, xb,DM,0, nullptr,0,
                  P(bF+5), xb, nullptr,0, TT,DM,DFF,1,0);
    }
  };

  for (int l = 0; l < 6; l++) { runMHA(l, false); runMHA(l, true); runFFN(l); }

  // ---- final LN + decoder + output assembly ----
  ln_kernel<<<(BTN+7)/8, 256, 0, stream>>>(x, h, P(180), P(181), BTN);
  launch_gemm(stream, h, DM,0, dcW1.p,dcW1.ld,0, nullptr,DM,0, tb,0, P(183),nullptr,P(184),2, BTN,DM,DM,1,0);
  launch_gemm(stream, tb,DM,0, dcW2.p,dcW2.ld,0, dec,257,0, nullptr,0, P(186),nullptr,nullptr,0, BTN,257,DM,1,0);

  float* mout  = (float*)d_out;
  float* score = mout + (size_t)BTN*256;
  finalize_kernel<<<blk((long long)BTN*257), 256, 0, stream>>>(dec, motion, mout, score);
}
